// SelfCorrelationBlock_50113678410732
// MI455X (gfx1250) — matrix operand order repaired and checked by reading, not compiled, not run on hardware
//
#include <hip/hip_runtime.h>
#include <hip/hip_bf16.h>

// ---------- problem dims ----------
#define C_IN 256
#define RDIM 64
#define HDIM 64
#define WDIM 64
#define BATCH 8
#define PIX_PER_IMG (HDIM * WDIM)       // 4096
#define NPIX (BATCH * PIX_PER_IMG)      // 32768
#define WG_PIX 128
#define WIN 7
#define PAD 3
#define TILE 16
#define HALO (TILE + 2 * PAD)           // 22
#define PADW (WDIM + 2 * PAD)           // 70 (padded feat width/height)
#define EPSF 1e-8f

typedef __attribute__((ext_vector_type(16))) __bf16 v16bf;
typedef __attribute__((ext_vector_type(8)))  float  v8f;

// ---------- TDM availability (device pass only; host pass takes fallback) ----------
#ifndef __has_builtin
#define __has_builtin(x) 0
#endif
#if defined(__HIP_DEVICE_COMPILE__) && __has_builtin(__builtin_amdgcn_tensor_load_to_lds)
#define HAVE_TDM 1
#else
#define HAVE_TDM 0
#endif

// ---------- bf16 helpers (native converts; RTNE) ----------
__device__ __forceinline__ unsigned short f2us(float f) {
    return __builtin_bit_cast(unsigned short, (__bf16)f);
}
__device__ __forceinline__ unsigned packbf2(float a, float b) {
    return (unsigned)__builtin_bit_cast(unsigned short, (__bf16)a) |
           ((unsigned)__builtin_bit_cast(unsigned short, (__bf16)b) << 16);
}
__device__ __forceinline__ float us2f(unsigned short s) {
    unsigned u = ((unsigned)s) << 16;
    return __builtin_bit_cast(float, u);
}
__device__ __forceinline__ __bf16 us2bf(unsigned short s) {
    return __builtin_bit_cast(__bf16, s);
}

#if HAVE_TDM
typedef unsigned int uint32x4 __attribute__((ext_vector_type(4)));
typedef int          int32x8  __attribute__((ext_vector_type(8)));
typedef int          int32x4  __attribute__((ext_vector_type(4)));

// 2-D TDM copy (dword elements) global -> LDS. tile_h==0 => 1-D (dim1 unused).
// D# layout per CDNA5 ISA.
__device__ __forceinline__ void tdm_load_2d_dwords(unsigned lds_off, const void* gptr,
                                                   unsigned tile_w, unsigned tile_h,
                                                   unsigned tensor_w, unsigned tensor_h,
                                                   unsigned stride_dw) {
    unsigned long long ga = (unsigned long long)(size_t)gptr;
    uint32x4 g0;
    g0[0] = 1u;                                                 // count=1, user mode
    g0[1] = lds_off;                                            // lds byte address
    g0[2] = (unsigned)(ga & 0xFFFFFFFFu);                       // global_addr[31:0]
    g0[3] = (unsigned)((ga >> 32) & 0x01FFFFFFu) | 0x80000000u; // addr[56:32] | type=2
    int32x8 g1;
    g1[0] = (int)(2u << 16);                                    // data_size = 4 bytes
    g1[1] = (int)((tensor_w & 0xFFFFu) << 16);                  // tensor_dim0 lo @bit48
    g1[2] = (int)((tensor_w >> 16) | ((tensor_h & 0xFFFFu) << 16));
    g1[3] = (int)((tensor_h >> 16) | ((tile_w & 0xFFFFu) << 16)); // tile_dim0 @bit112
    g1[4] = (int)(tile_h & 0xFFFFu);                            // tile_dim1 (dim2=0)
    g1[5] = (int)stride_dw;                                     // tensor_dim0_stride lo
    g1[6] = 0;
    g1[7] = 0;
    int32x4 z4 = {0, 0, 0, 0};
#if __clang_major__ >= 23
    int32x8 z8 = {0, 0, 0, 0, 0, 0, 0, 0};
    __builtin_amdgcn_tensor_load_to_lds(g0, g1, z4, z4, z8, 0);
#else
    __builtin_amdgcn_tensor_load_to_lds(g0, g1, z4, z4, 0);
#endif
}
#endif // HAVE_TDM

// ---------------------------------------------------------------------------
// Kernel Z: zero the padded feat buffer (borders must be 0 for window kernel)
// ---------------------------------------------------------------------------
__global__ void zero_feat_kernel(uint4* __restrict__ p, int n4) {
    uint4 z = {0u, 0u, 0u, 0u};
    int stride = gridDim.x * blockDim.x;
    for (int i = blockIdx.x * blockDim.x + threadIdx.x; i < n4; i += stride) p[i] = z;
}

// ---------------------------------------------------------------------------
// Kernel 0: convert W_reduce (64x256) and W_proj (256x64) f32 -> bf16 in ws
// ---------------------------------------------------------------------------
__global__ void cvt_weights_kernel(const float* __restrict__ wred_f,
                                   const float* __restrict__ wproj_f,
                                   unsigned short* __restrict__ wred_b,
                                   unsigned short* __restrict__ wproj_b) {
    int i = blockIdx.x * 256 + threadIdx.x;
    if (i < RDIM * C_IN)  wred_b[i]  = f2us(wred_f[i]);
    if (i < C_IN * RDIM)  wproj_b[i] = f2us(wproj_f[i]);
}

// ---------------------------------------------------------------------------
// Kernel 1: feat_p[b][y+3][x+3][r] = sum_c W_reduce[r][c]*x[b][c][y][x]
// bf16 WMMA GEMM. W_reduce staged once via 1-D TDM; x chunks (32c x 128pix f32)
// streamed through a double-buffered 2-D TDM pipeline (wave 0 issues,
// s_wait_tensorcnt<=1 overlaps DMA of chunk s+1 with compute of chunk s).
// ---------------------------------------------------------------------------
#define XBUF0 ((unsigned)(RDIM * C_IN * 2))     // 32768: f32 buffer 0 (16 KB)
#define XBUFSZ 16384u
__global__ void feat_gemm_kernel(const float* __restrict__ x,
                                 const unsigned short* __restrict__ Wred,
                                 unsigned short* __restrict__ feat_p) {
    extern __shared__ char smem_raw[];
    unsigned short* Wlds = (unsigned short*)smem_raw;            // [64][256] bf16

    const int tid  = threadIdx.x;
    const int wave = tid >> 5;
    const int lane = tid & 31;
    const int p0   = blockIdx.x * WG_PIX;
    const int b    = p0 / PIX_PER_IMG;
    const int hw0  = p0 % PIX_PER_IMG;
    const float* xb = x + (size_t)b * C_IN * PIX_PER_IMG + hw0;  // tile pixel col 0

#if HAVE_TDM
    if (tid < 32) {                                              // wave 0 drives DMA
        tdm_load_2d_dwords(0u, Wred, (RDIM * C_IN) / 2, 0,
                           (RDIM * C_IN) / 2, 1, (RDIM * C_IN) / 2);
        tdm_load_2d_dwords(XBUF0, xb, WG_PIX, 32,
                           PIX_PER_IMG, C_IN, PIX_PER_IMG);      // chunk 0
    }
#else
    for (int i = tid; i < RDIM * C_IN; i += 256) Wlds[i] = Wred[i];
#endif

    const int n  = lane & 15;
    const int hi = lane >> 4;
    const int mypix = wave * 16 + n;

    v8f acc[4];
#pragma unroll
    for (int mt = 0; mt < 4; ++mt)
#pragma unroll
        for (int v = 0; v < 8; ++v) acc[mt][v] = 0.0f;

    for (int step = 0; step < 8; ++step) {
        const int ko = step * 32;
        const unsigned cur_off = XBUF0 + (unsigned)(step & 1) * XBUFSZ;
#if HAVE_TDM
        if (tid < 32) {
            if (step < 7) {
                tdm_load_2d_dwords(XBUF0 + (unsigned)((step + 1) & 1) * XBUFSZ,
                                   xb + (size_t)(ko + 32) * PIX_PER_IMG,
                                   WG_PIX, 32, PIX_PER_IMG, C_IN, PIX_PER_IMG);
                __builtin_amdgcn_s_wait_tensorcnt(1);   // chunk `step` landed
            } else {
                __builtin_amdgcn_s_wait_tensorcnt(0);   // last chunk landed
            }
        }
#else
        {
            float* Xw = (float*)(smem_raw + cur_off);
            for (int j = 0; j < 16; ++j) {
                int i = j * 256 + tid;
                int cc = i >> 7, pp = i & 127;
                Xw[cc * WG_PIX + pp] =
                    x[((size_t)(b * C_IN + ko + cc)) * PIX_PER_IMG + hw0 + pp];
            }
        }
#endif
        __syncthreads();                        // chunk `step` visible to all waves
        const float* Xf = (const float*)(smem_raw + cur_off);   // [32 c][128 pix] f32

        // B fragment: n = lane&15, k = hi*16 + e  (convert f32->bf16 in regs)
        v16bf bfrag;
#pragma unroll
        for (int e = 0; e < 16; ++e)
            bfrag[e] = (__bf16)Xf[(hi * 16 + e) * WG_PIX + mypix];

#pragma unroll
        for (int mt = 0; mt < 4; ++mt) {
            // A fragment: m = lane&15, k = hi*16 + e
            v16bf afrag;
            const int m = mt * 16 + n;
#pragma unroll
            for (int e = 0; e < 16; ++e) {
                int k = hi * 16 + e;
                afrag[e] = us2bf(Wlds[m * C_IN + ko + k]);
            }
            acc[mt] = __builtin_amdgcn_wmma_f32_16x16x32_bf16(
                false, afrag, false, bfrag, (short)0, acc[mt], false, false);
        }
        __syncthreads();                        // buffer consumed; DMA may overwrite
    }

    // store feat bf16 pairs into padded [b][70][70][64] layout (D: m = v + hi*8)
    const int hw = hw0 + mypix;
    const int yy = (hw >> 6) + PAD;
    const int xx = (hw & 63) + PAD;
    unsigned short* fp = feat_p + ((size_t)((b * PADW + yy) * PADW + xx)) * RDIM;
#pragma unroll
    for (int mt = 0; mt < 4; ++mt)
#pragma unroll
        for (int v = 0; v < 8; v += 2) {
            int r = mt * 16 + v + hi * 8;       // even
            *(unsigned*)(fp + r) = packbf2(acc[mt][v], acc[mt][v + 1]);
        }
}

// ---------------------------------------------------------------------------
// Kernel 2: 7x7 window cosine-similarity attention out of an LDS halo tile.
// Padded feat buffer => halo = one 2-D TDM copy (22 rows x 704 dwords), no
// bounds logic at all. WG = 256 thr = 16x16 pixel tile.
// ---------------------------------------------------------------------------
__global__ void window_attn_kernel(const unsigned short* __restrict__ feat_p,
                                   unsigned short* __restrict__ corr) {
    extern __shared__ char smem_raw[];
    unsigned short* F = (unsigned short*)smem_raw;      // [HALO*HALO][64]

    const int tid = threadIdx.x;
    const int wg  = blockIdx.x;
    const int b   = wg >> 4;
    const int t   = wg & 15;
    const int h0  = (t >> 2) * TILE;                    // halo start == h0 in padded coords
    const int w0  = (t & 3) * TILE;

    const unsigned short* src =
        feat_p + ((size_t)((b * PADW + h0) * PADW + w0)) * RDIM;
#if HAVE_TDM
    if (tid < 32) {
        tdm_load_2d_dwords(0u, src,
                           (HALO * RDIM) / 2, HALO,               // 704 dw x 22 rows
                           (PADW * RDIM) / 2, PADW,               // tensor row = 2240 dw
                           (PADW * RDIM) / 2);
        __builtin_amdgcn_s_wait_tensorcnt(0);
    }
#else
    {
        const unsigned int* s32 = (const unsigned int*)src;
        unsigned int* d32 = (unsigned int*)F;
        const int ROWDW = (HALO * RDIM) / 2;            // 704
        for (int i = tid; i < HALO * ROWDW; i += 256) {
            int row = i / ROWDW, cd = i % ROWDW;
            d32[i] = s32[(size_t)row * ((PADW * RDIM) / 2) + cd];
        }
    }
#endif
    __syncthreads();

    const int ly = tid >> 4, lx = tid & 15;
    const unsigned short* cv = &F[((ly + PAD) * HALO + (lx + PAD)) * RDIM];

    float cn2 = 0.0f;
#pragma unroll 8
    for (int r = 0; r < RDIM; ++r) { float a = us2f(cv[r]); cn2 += a * a; }
    const float cn = fmaxf(__builtin_sqrtf(cn2), EPSF);

    float sims[WIN * WIN];
    int k = 0;
#pragma unroll
    for (int dy = 0; dy < WIN; ++dy)
#pragma unroll
        for (int dx = 0; dx < WIN; ++dx) {
            const unsigned short* pv = &F[((ly + dy) * HALO + (lx + dx)) * RDIM];
            float dot = 0.0f, pn2 = 0.0f;
#pragma unroll 8
            for (int r = 0; r < RDIM; ++r) {
                float a = us2f(cv[r]);
                float p = us2f(pv[r]);
                dot += a * p;
                pn2 += p * p;
            }
            sims[k++] = dot / (cn * fmaxf(__builtin_sqrtf(pn2), EPSF));
        }

    float mx = -1e30f;
#pragma unroll
    for (int i = 0; i < WIN * WIN; ++i) mx = fmaxf(mx, sims[i]);
    float ssum = 0.0f;
#pragma unroll
    for (int i = 0; i < WIN * WIN; ++i) { sims[i] = __expf(sims[i] - mx); ssum += sims[i]; }
    const float inv = 1.0f / ssum;

    float accR[RDIM];
#pragma unroll
    for (int r = 0; r < RDIM; ++r) accR[r] = 0.0f;
    k = 0;
#pragma unroll
    for (int dy = 0; dy < WIN; ++dy)
#pragma unroll
        for (int dx = 0; dx < WIN; ++dx) {
            const float a = sims[k++] * inv;
            const unsigned short* pv = &F[((ly + dy) * HALO + (lx + dx)) * RDIM];
#pragma unroll 8
            for (int r = 0; r < RDIM; ++r) accR[r] += a * us2f(pv[r]);
        }

    const size_t pix = (size_t)b * PIX_PER_IMG + (h0 + ly) * WDIM + (w0 + lx);
    unsigned short* cp = corr + pix * RDIM;
#pragma unroll
    for (int r = 0; r < RDIM; r += 2)
        *(unsigned*)(cp + r) = packbf2(accR[r], accR[r + 1]);
}

// ---------------------------------------------------------------------------
// Kernel 3: out = x + W_proj @ corr   (bf16 WMMA GEMM + fused residual)
// W_proj (8192 dw) and corr tile (4096 dw) staged via 1-D TDM copies.
// ---------------------------------------------------------------------------
__global__ void proj_gemm_kernel(const float* __restrict__ x,
                                 const unsigned short* __restrict__ Wproj,
                                 const unsigned short* __restrict__ corr,
                                 float* __restrict__ out) {
    extern __shared__ char smem_raw[];
    unsigned short* Wlds = (unsigned short*)smem_raw;                        // [256][64]
    unsigned short* Clds = (unsigned short*)(smem_raw + C_IN * RDIM * 2);    // [128][64]

    const int tid  = threadIdx.x;
    const int wave = tid >> 5;
    const int lane = tid & 31;
    const int p0   = blockIdx.x * WG_PIX;
    const int b    = p0 / PIX_PER_IMG;
    const int hw0  = p0 % PIX_PER_IMG;

#if HAVE_TDM
    if (tid < 32) {
        tdm_load_2d_dwords(0u, Wproj, (C_IN * RDIM) / 2, 0,
                           (C_IN * RDIM) / 2, 1, (C_IN * RDIM) / 2);
        tdm_load_2d_dwords((unsigned)(C_IN * RDIM * 2), corr + (size_t)p0 * RDIM,
                           (WG_PIX * RDIM) / 2, 0,
                           (WG_PIX * RDIM) / 2, 1, (WG_PIX * RDIM) / 2);
        __builtin_amdgcn_s_wait_tensorcnt(0);
    }
#else
    for (int i = tid; i < C_IN * RDIM; i += 256) Wlds[i] = Wproj[i];
    {
        const unsigned int* src = (const unsigned int*)(corr + (size_t)p0 * RDIM);
        unsigned int* dst = (unsigned int*)Clds;
        for (int i = tid; i < (WG_PIX * RDIM) / 2; i += 256) dst[i] = src[i];
    }
#endif
    __syncthreads();

    const int n  = lane & 15;
    const int hi = lane >> 4;
    const int mypix = wave * 16 + n;

    v16bf bfrag[2];
#pragma unroll
    for (int kb = 0; kb < 2; ++kb)
#pragma unroll
        for (int e = 0; e < 16; ++e) {
            int k = kb * 32 + hi * 16 + e;     // contiguous per lane
            bfrag[kb][e] = us2bf(Clds[mypix * RDIM + k]);
        }

    const int hw = hw0 + mypix;
#pragma unroll
    for (int mt = 0; mt < 16; ++mt) {
        v8f acc;
#pragma unroll
        for (int v = 0; v < 8; ++v) acc[v] = 0.0f;
#pragma unroll
        for (int kb = 0; kb < 2; ++kb) {
            v16bf afrag;
            const int m = mt * 16 + n;
#pragma unroll
            for (int e = 0; e < 16; ++e) {
                int k = kb * 32 + hi * 16 + e;
                afrag[e] = us2bf(Wlds[m * RDIM + k]);
            }
            acc = __builtin_amdgcn_wmma_f32_16x16x32_bf16(
                false, afrag, false, bfrag[kb], (short)0, acc, false, false);
        }
#pragma unroll
        for (int v = 0; v < 8; ++v) {
            int cout = mt * 16 + v + hi * 8;
            size_t gidx = ((size_t)(b * C_IN + cout)) * PIX_PER_IMG + hw;
            out[gidx] = x[gidx] + acc[v];
        }
    }
}

// ---------------------------------------------------------------------------
extern "C" void kernel_launch(void* const* d_in, const int* in_sizes, int n_in,
                              void* d_out, int out_size, void* d_ws, size_t ws_size,
                              hipStream_t stream) {
    (void)in_sizes; (void)n_in; (void)out_size; (void)ws_size;
    const float* x     = (const float*)d_in[0];
    const float* wred  = (const float*)d_in[1];
    const float* wproj = (const float*)d_in[2];
    float* out = (float*)d_out;

    char* ws = (char*)d_ws;
    unsigned short* wred_b  = (unsigned short*)ws;                          // 32 KB
    unsigned short* wproj_b = (unsigned short*)(ws + 32768);                // 32 KB
    unsigned short* feat_p  = (unsigned short*)(ws + 65536);                // 70*70*64*2*8 B
    const size_t feat_bytes = (size_t)BATCH * PADW * PADW * RDIM * 2;       // 5,017,600
    unsigned short* corr    = (unsigned short*)(ws + 65536 + feat_bytes);   // 4 MB

    zero_feat_kernel<<<512, 256, 0, stream>>>((uint4*)feat_p, (int)(feat_bytes / 16));
    cvt_weights_kernel<<<64, 256, 0, stream>>>(wred, wproj, wred_b, wproj_b);

    const int feat_lds = RDIM * C_IN * 2 + 2 * (int)XBUFSZ;                 // 65536 B
    feat_gemm_kernel<<<NPIX / WG_PIX, 256, feat_lds, stream>>>(x, wred_b, feat_p);

    const int attn_lds = HALO * HALO * RDIM * 2;                            // 61952 B
    window_attn_kernel<<<BATCH * 16, 256, attn_lds, stream>>>(feat_p, corr);

    const int proj_lds = C_IN * RDIM * 2 + WG_PIX * RDIM * 2;               // 49152 B
    proj_gemm_kernel<<<NPIX / WG_PIX, 256, proj_lds, stream>>>(x, wproj_b, corr, out);
}
